// AdvancedPatchContrastiveLoss_71665824301676
// MI455X (gfx1250) — matrix operand order repaired and checked by reading, not compiled, not run on hardware
//
#include <hip/hip_runtime.h>
#include <math.h>

// ---------------------------------------------------------------------------
// Problem constants (B=32, L=256, D=256 from the reference setup)
// ---------------------------------------------------------------------------
#define N_TOK 8192           // B*L
#define D_DIM 256
#define NB    4096           // histogram bins per refinement level
#define TILE  128            // block output tile is TILE x TILE
#define LDA   40             // padded LDS row stride in halves (bank-conflict free)
#define TILE_BYTES (TILE * LDA * 2)   // 10240 bytes per staged K-slab
#define INV_T (1.0f/0.07f)
#define EPS_F 1e-8f
#define LO1   (-14.5f)       // |sim| <= 1/0.07 ~= 14.2857
#define RANGE1 29.0f

typedef _Float16 v16h __attribute__((ext_vector_type(16)));
typedef _Float16 v8h  __attribute__((ext_vector_type(8)));
typedef float    v8f  __attribute__((ext_vector_type(8)));
typedef unsigned int v4u __attribute__((ext_vector_type(4)));
typedef int          v8i __attribute__((ext_vector_type(8)));
typedef int          v4i __attribute__((ext_vector_type(4)));

struct Ctl {
    unsigned long long dcount[4];     // per-dataset counts
    long long M;                      // # cross-dataset pairs
    long long k;                      // floor(0.8*(M-1))
    long long C0;                     // cumulative count below refine window
    float frac;                       // fractional rank
    float lo1, binw1;                 // pass-1 histogram params
    float lo2, binw2;                 // pass-2 (refined) histogram params
    float thr;                        // final quantile threshold
};

// ---------------------------------------------------------------------------
// Utility kernels
// ---------------------------------------------------------------------------
__global__ void zero_kernel(unsigned* p, int n) {
    int i = blockIdx.x * 256 + threadIdx.x;
    if (i < n) p[i] = 0u;
}

__global__ void count_kernel(const int* __restrict__ did, Ctl* ctl) {
    __shared__ unsigned c4[4];
    int t = threadIdx.x;
    if (t < 4) c4[t] = 0u;
    __syncthreads();
    int i = blockIdx.x * 256 + t;
    atomicAdd(&c4[did[i] & 3], 1u);
    __syncthreads();
    if (t < 4) atomicAdd(&ctl->dcount[t], (unsigned long long)c4[t]);
}

// One wave normalizes one row of 256 floats and emits f16.
__global__ void __launch_bounds__(256) normalize_kernel(const float* __restrict__ feats,
                                                        _Float16* __restrict__ Xh) {
    int t = threadIdx.x;
    int wave = t >> 5, lane = t & 31;
    int row = blockIdx.x * 8 + wave;
    const float4* src = reinterpret_cast<const float4*>(feats + (size_t)row * D_DIM);
    float4 a = src[lane * 2];
    float4 b = src[lane * 2 + 1];
    float ss = a.x*a.x + a.y*a.y + a.z*a.z + a.w*a.w
             + b.x*b.x + b.y*b.y + b.z*b.z + b.w*b.w;
#pragma unroll
    for (int m = 16; m >= 1; m >>= 1) ss += __shfl_xor(ss, m, 32);
    float inv = 1.0f / fmaxf(sqrtf(ss), EPS_F);
    v8h h;
    h[0] = (_Float16)(a.x * inv); h[1] = (_Float16)(a.y * inv);
    h[2] = (_Float16)(a.z * inv); h[3] = (_Float16)(a.w * inv);
    h[4] = (_Float16)(b.x * inv); h[5] = (_Float16)(b.y * inv);
    h[6] = (_Float16)(b.z * inv); h[7] = (_Float16)(b.w * inv);
    ((v8h*)(Xh + (size_t)row * D_DIM))[lane] = h;
}

__global__ void prep_kernel(Ctl* ctl) {
    long long s2 = 0;
    for (int d = 0; d < 4; ++d) {
        long long c = (long long)ctl->dcount[d];
        s2 += c * c;
    }
    long long M = (long long)N_TOK * (long long)N_TOK - s2;
    ctl->M = M;
    if (M > 0) {
        double pos = 0.8 * (double)(M - 1);
        long long k = (long long)pos;           // pos >= 0 -> floor
        ctl->k = k;
        ctl->frac = (float)(pos - (double)k);
    } else {
        ctl->k = 0; ctl->frac = 0.0f;
    }
    ctl->lo1 = LO1;
    ctl->binw1 = RANGE1 / (float)NB;
    ctl->thr = 0.0f;
}

// ---------------------------------------------------------------------------
// TDM: async DMA of one 128x32-halfword tile (row-major, stride 256 halves)
// into LDS at byte offset lds_off, with TDM-injected padding so LDS rows land
// at a 40-half (80 B) stride: 16 DWORDs written, then 4 DWORDs of pad.
// D# layout per CDNA5 ISA sec 8.3/8.4. Tracked by TENSORcnt.
// ---------------------------------------------------------------------------
__device__ __forceinline__ void tdm_load_tile(unsigned lds_off,
                                              const _Float16* gtile) {
    unsigned long long ga = (unsigned long long)(size_t)gtile;
    v4u g0;
    g0.x = 1u;                                           // count=1 (valid), user mode
    g0.y = lds_off;                                      // lds_addr (bytes)
    g0.z = (unsigned)(ga & 0xFFFFFFFFull);               // global_addr[31:0]
    g0.w = (unsigned)((ga >> 32) & 0x1FFFFFFull)         // global_addr[56:32]
         | (2u << 30);                                   // type = 2 ("image")
    v8i g1;
    g1[0] = (int)((1u << 16)        // data_size = 1 -> 2 bytes
                | (1u << 20)        // pad_enable
                | (3u << 22)        // pad_interval = 3 -> 16 DWORDs before pad
                | (3u << 25));      // pad_amount   = 3 -> 4 DWORDs of pad
    g1[1] = (int)(256u << 16);      // tensor_dim0[15:0]=256 (atomic_barrier_addr=0)
    g1[2] = (int)(128u << 16);      // tensor_dim0[31:16]=0 | tensor_dim1[15:0]=128
    g1[3] = (int)(32u  << 16);      // tensor_dim1[31:16]=0 | tile_dim0=32
    g1[4] = (int)(128u);            // tile_dim1=128 | tile_dim2=0
    g1[5] = (int)(256u);            // tensor_dim0_stride[31:0]=256
    g1[6] = 0;                      // stride0[47:32]=0 | stride1[15:0]=0
    g1[7] = 0;                      // tensor_dim1_stride[47:16]=0
    v4i gz4 = {0, 0, 0, 0};         // groups 2/3 unused for 2-D tile
    v8i gz8 = {0, 0, 0, 0, 0, 0, 0, 0};
    __builtin_amdgcn_tensor_load_to_lds(g0, g1, gz4, gz4, gz8, 0);
}

// ---------------------------------------------------------------------------
// Core 128x128 tile GEMM via WMMA f32 <- f16 x f16 (wave32, 16x16x32).
// Double-buffered LDS K-slabs fetched by the Tensor Data Mover so that the
// DMA for step kb+1 overlaps the 8 WMMAs of step kb. smem layout:
//   [0*TB) A buf0 | [1*TB) B buf0 | [2*TB) A buf1 | [3*TB) B buf1 | user...
// ---------------------------------------------------------------------------
__device__ __forceinline__ void gemm_tile(const _Float16* __restrict__ Xh,
                                          int I0, int J0,
                                          char* smem, v8f acc[8]) {
    const int t = threadIdx.x;
    const int w = t >> 5, lane = t & 31;
    const int m = lane & 15, hi = lane >> 4;

#pragma unroll
    for (int n = 0; n < 8; ++n)
#pragma unroll
        for (int e = 0; e < 8; ++e) acc[n][e] = 0.0f;

    // Prologue: wave 0 kicks off DMA of K-slab 0 into buffer 0.
    if (w == 0) {
        tdm_load_tile(0u * TILE_BYTES, Xh + (size_t)I0 * D_DIM);
        tdm_load_tile(1u * TILE_BYTES, Xh + (size_t)J0 * D_DIM);
    }

    for (int kb = 0; kb < 8; ++kb) {
        const int cur = kb & 1;
        if (w == 0) __builtin_amdgcn_s_wait_tensorcnt(0);
        __syncthreads();   // slab `cur` resident; all waves done reading `cur^1`
        if (w == 0 && kb < 7) {
            const int nxt = cur ^ 1;
            tdm_load_tile((unsigned)(2 * nxt + 0) * TILE_BYTES,
                          Xh + (size_t)I0 * D_DIM + (kb + 1) * 32);
            tdm_load_tile((unsigned)(2 * nxt + 1) * TILE_BYTES,
                          Xh + (size_t)J0 * D_DIM + (kb + 1) * 32);
        }
        const _Float16* As = (const _Float16*)(smem + (2 * cur + 0) * TILE_BYTES);
        const _Float16* Bs = (const _Float16*)(smem + (2 * cur + 1) * TILE_BYTES);

        // A fragment: lane<16 holds K {0..7,16..23}, lane>=16 holds {8..15,24..31}.
        const _Float16* arow = As + (w * 16 + m) * LDA;
        v8h a0 = *(const v8h*)(arow + hi * 8);
        v8h a1 = *(const v8h*)(arow + 16 + hi * 8);
        v16h af;
#pragma unroll
        for (int e = 0; e < 8; ++e) { af[e] = a0[e]; af[8 + e] = a1[e]; }

#pragma unroll
        for (int n = 0; n < 8; ++n) {
            // B fragment: lane holds column N=m, K {0..7,16..23} (lane<16) or {8..15,24..31}.
            const _Float16* brow = Bs + (n * 16 + m) * LDA + hi * 8;
            v8h b0 = *(const v8h*)(brow);
            v8h b1 = *(const v8h*)(brow + 16);
            v16h bf;
#pragma unroll
            for (int e = 0; e < 8; ++e) { bf[e] = b0[e]; bf[8 + e] = b1[e]; }
            acc[n] = __builtin_amdgcn_wmma_f32_16x16x32_f16(
                false, af, false, bf, (short)0, acc[n], false, false);
        }
    }
}

// Dynamic-LDS layout offsets (tiles occupy [0, 4*TILE_BYTES))
#define SMEM_TILES (4 * TILE_BYTES)                 // 40960
#define HIST_SMEM_BYTES (SMEM_TILES + NB * 4 + 2 * TILE * 4)   // 58368
#define LOSS_SMEM_BYTES (SMEM_TILES + 4 * TILE * 4)            // 43008

// ---------------------------------------------------------------------------
// Pass 1/2: GEMM + histogram of cross-dataset similarities
// ---------------------------------------------------------------------------
__global__ void __launch_bounds__(256) gemm_hist_kernel(const _Float16* __restrict__ Xh,
                                                        const int* __restrict__ did,
                                                        const Ctl* __restrict__ ctl,
                                                        unsigned* __restrict__ ghist,
                                                        int pass) {
    extern __shared__ char smem[];
    unsigned* hist_s = (unsigned*)(smem + SMEM_TILES);
    int* didI = (int*)(smem + SMEM_TILES + NB * 4);
    int* didJ = didI + TILE;

    int t = threadIdx.x;
    int I0 = blockIdx.x * TILE, J0 = blockIdx.y * TILE;
    for (int b = t; b < NB; b += 256) hist_s[b] = 0u;
    if (t < 128) didI[t] = did[I0 + t];
    else         didJ[t - 128] = did[J0 + t - 128];

    v8f acc[8];
    gemm_tile(Xh, I0, J0, smem, acc);   // contains the needed barriers

    float lo   = (pass == 1) ? ctl->lo1   : ctl->lo2;
    float binw = (pass == 1) ? ctl->binw1 : ctl->binw2;
    float invb = 1.0f / binw;

    int w = t >> 5, lane = t & 31, m = lane & 15, hi = lane >> 4;
#pragma unroll
    for (int n = 0; n < 8; ++n) {
        int dj = didJ[n * 16 + m];
#pragma unroll
        for (int r = 0; r < 8; ++r) {
            int di = didI[w * 16 + r + 8 * hi];
            if (di != dj) {
                float sim = acc[n][r] * INV_T;
                int b = (int)floorf((sim - lo) * invb);
                if (pass == 1) {
                    b = b < 0 ? 0 : (b > NB - 1 ? NB - 1 : b);
                    atomicAdd(&hist_s[b], 1u);
                } else if (b >= 0 && b < NB) {
                    atomicAdd(&hist_s[b], 1u);
                }
            }
        }
    }
    __syncthreads();
    for (int b = t; b < NB; b += 256) {
        unsigned c = hist_s[b];
        if (c) atomicAdd(&ghist[b], c);
    }
}

// rank -> bin lookup using per-thread segment partials; returns count below bin
__device__ long long find_bin(const unsigned* ghist, const unsigned long long* part,
                              long long rank, int* bin_out) {
    long long cum = 0;
    int sg = 0;
    while (sg < 256 && cum + (long long)part[sg] <= rank) { cum += part[sg]; ++sg; }
    if (sg == 256) { *bin_out = NB - 1; return cum; }
    int b = sg * (NB / 256);
    int bend = b + (NB / 256);
    while (b < bend && cum + (long long)ghist[b] <= rank) { cum += ghist[b]; ++b; }
    if (b == bend) b = bend - 1;
    *bin_out = b;
    return cum;
}

__global__ void select_kernel(Ctl* ctl, unsigned* ghist) {
    __shared__ unsigned long long part[256];
    int t = threadIdx.x;
    unsigned long long s = 0;
    for (int j = 0; j < NB / 256; ++j) s += ghist[t * (NB / 256) + j];
    part[t] = s;
    __syncthreads();
    if (t == 0) {
        long long M = ctl->M;
        if (M > 0) {
            long long k = ctl->k;
            long long k1 = (k + 1 < M) ? (k + 1) : (M - 1);
            int bk, bk1;
            long long c0 = find_bin(ghist, part, k, &bk);
            find_bin(ghist, part, k1, &bk1);
            float lo2 = ctl->lo1 + (float)bk * ctl->binw1;
            float hi2 = ctl->lo1 + (float)(bk1 + 1) * ctl->binw1;
            ctl->lo2 = lo2;
            ctl->binw2 = (hi2 - lo2) / (float)NB;
            ctl->C0 = c0;
        } else {
            ctl->lo2 = 0.0f; ctl->binw2 = 1.0f; ctl->C0 = 0;
        }
    }
    __syncthreads();
    for (int b = t; b < NB; b += 256) ghist[b] = 0u;   // reuse buffer for pass 2
}

__global__ void thr_kernel(Ctl* ctl, unsigned* ghist) {
    __shared__ unsigned long long part[256];
    int t = threadIdx.x;
    unsigned long long s = 0;
    for (int j = 0; j < NB / 256; ++j) s += ghist[t * (NB / 256) + j];
    part[t] = s;
    __syncthreads();
    if (t == 0) {
        long long M = ctl->M;
        if (M > 0) {
            long long k = ctl->k;
            long long k1 = (k + 1 < M) ? (k + 1) : (M - 1);
            long long r0 = k - ctl->C0;  if (r0 < 0) r0 = 0;
            long long r1 = k1 - ctl->C0; if (r1 < r0) r1 = r0;
            int b0, b1;
            find_bin(ghist, part, r0, &b0);
            find_bin(ghist, part, r1, &b1);
            float v0 = ctl->lo2 + ((float)b0 + 0.5f) * ctl->binw2;
            float v1 = ctl->lo2 + ((float)b1 + 0.5f) * ctl->binw2;
            ctl->thr = v0 + ctl->frac * (v1 - v0);
        } else {
            ctl->thr = 0.0f;
        }
    }
}

// ---------------------------------------------------------------------------
// Pass 3: GEMM + per-row weighted pos/neg exp-sums
// ---------------------------------------------------------------------------
__global__ void __launch_bounds__(256) gemm_loss_kernel(const _Float16* __restrict__ Xh,
                                                        const int* __restrict__ did,
                                                        const int* __restrict__ iid,
                                                        const Ctl* __restrict__ ctl,
                                                        float* __restrict__ pos_sum,
                                                        float* __restrict__ neg_sum) {
    extern __shared__ char smem[];
    int* didI = (int*)(smem + SMEM_TILES);
    int* didJ = didI + TILE;
    int* iidI = didJ + TILE;
    int* iidJ = iidI + TILE;

    int t = threadIdx.x;
    int I0 = blockIdx.x * TILE, J0 = blockIdx.y * TILE;
    if (t < 128) { didI[t] = did[I0 + t]; iidI[t] = iid[I0 + t]; }
    else { int u = t - 128; didJ[u] = did[J0 + u]; iidJ[u] = iid[J0 + u]; }

    v8f acc[8];
    gemm_tile(Xh, I0, J0, smem, acc);

    float thr = ctl->thr;
    int w = t >> 5, lane = t & 31, m = lane & 15, hi = lane >> 4;

    float pos_p[8], neg_p[8];
#pragma unroll
    for (int r = 0; r < 8; ++r) { pos_p[r] = 0.0f; neg_p[r] = 0.0f; }

#pragma unroll
    for (int n = 0; n < 8; ++n) {
        int jl = n * 16 + m;
        int dj = didJ[jl], ij = iidJ[jl];
        int j = J0 + jl;
#pragma unroll
        for (int r = 0; r < 8; ++r) {
            int il = w * 16 + r + 8 * hi;
            int di = didI[il], ii = iidI[il];
            int i = I0 + il;
            float sim = acc[n][r] * INV_T;
            float e = expf(sim);
            bool same = (di == dj) && (ii == ij);
            if (same && (i != j)) {
                pos_p[r] += e;
            } else {
                float wgt = 1.0f;
                if (di != dj) wgt = 1.5f * ((sim > thr) ? 2.0f : 1.0f);
                neg_p[r] += wgt * e;
            }
        }
    }

    // Reduce across the 16 lanes of each half-wave (same row i), then atomics.
#pragma unroll
    for (int r = 0; r < 8; ++r) {
        float p = pos_p[r], q = neg_p[r];
#pragma unroll
        for (int mk = 8; mk >= 1; mk >>= 1) {
            p += __shfl_xor(p, mk, 32);
            q += __shfl_xor(q, mk, 32);
        }
        if (m == 0) {   // lanes 0 and 16 -> rows r and r+8 of this wave's strip
            int i = I0 + w * 16 + r + 8 * hi;
            atomicAdd(&pos_sum[i], p);
            atomicAdd(&neg_sum[i], q);
        }
    }
}

__global__ void finalize_kernel(const float* __restrict__ pos,
                                const float* __restrict__ neg,
                                float* __restrict__ out) {
    __shared__ float sv[256], sa[256];
    __shared__ unsigned sn[256];
    int t = threadIdx.x;
    float lv = 0.0f, la = 0.0f;
    unsigned nv = 0u;
    for (int i = t; i < N_TOK; i += 256) {
        float p = pos[i], q = neg[i];
        float l = -logf((p + EPS_F) / (p + q + EPS_F));
        la += l;
        if (p > 0.0f) { lv += l; ++nv; }
    }
    sv[t] = lv; sa[t] = la; sn[t] = nv;
    __syncthreads();
    for (int s = 128; s > 0; s >>= 1) {
        if (t < s) { sv[t] += sv[t + s]; sa[t] += sa[t + s]; sn[t] += sn[t + s]; }
        __syncthreads();
    }
    if (t == 0) out[0] = (sn[0] > 0u) ? (sv[0] / (float)sn[0]) : (sa[0] / (float)N_TOK);
}

// ---------------------------------------------------------------------------
// Launch
// ---------------------------------------------------------------------------
extern "C" void kernel_launch(void* const* d_in, const int* in_sizes, int n_in,
                              void* d_out, int out_size, void* d_ws, size_t ws_size,
                              hipStream_t stream) {
    const float* feats = (const float*)d_in[0];
    const int* did = (const int*)d_in[1];
    const int* iid = (const int*)d_in[2];
    float* out = (float*)d_out;

    char* w = (char*)d_ws;
    _Float16* Xh = (_Float16*)w;          w += (size_t)N_TOK * D_DIM * sizeof(_Float16);
    Ctl* ctl = (Ctl*)w;                   w += sizeof(Ctl);
    unsigned* ghist = (unsigned*)w;       w += (size_t)NB * sizeof(unsigned);
    float* pos_sum = (float*)w;           w += (size_t)N_TOK * sizeof(float);
    float* neg_sum = (float*)w;

    // Zero Ctl + ghist + pos_sum + neg_sum (contiguous dword region).
    int nz = (int)(sizeof(Ctl) / 4) + NB + 2 * N_TOK;
    zero_kernel<<<(nz + 255) / 256, 256, 0, stream>>>((unsigned*)ctl, nz);

    count_kernel<<<N_TOK / 256, 256, 0, stream>>>(did, ctl);
    normalize_kernel<<<N_TOK / 8, 256, 0, stream>>>(feats, Xh);
    prep_kernel<<<1, 1, 0, stream>>>(ctl);

    dim3 grid(N_TOK / TILE, N_TOK / TILE);   // 64 x 64
    gemm_hist_kernel<<<grid, 256, HIST_SMEM_BYTES, stream>>>(Xh, did, ctl, ghist, 1);
    select_kernel<<<1, 256, 0, stream>>>(ctl, ghist);
    gemm_hist_kernel<<<grid, 256, HIST_SMEM_BYTES, stream>>>(Xh, did, ctl, ghist, 2);
    thr_kernel<<<1, 256, 0, stream>>>(ctl, ghist);
    gemm_loss_kernel<<<grid, 256, LOSS_SMEM_BYTES, stream>>>(Xh, did, iid, ctl, pos_sum, neg_sum);
    finalize_kernel<<<1, 256, 0, stream>>>(pos_sum, neg_sum, out);
}
